// IIDSegmentationLoss_55740085567577
// MI455X (gfx1250) — matrix operand order repaired and checked by reading, not compiled, not run on hardware
//
#include <hip/hip_runtime.h>

// IID segmentation loss on MI455X (gfx1250, wave32, WMMA + TDM).
// Stage 0: pack f32 -> f16 with spatial halo baked in (zero padding in global).
// Stage 1: 225-shift cross-correlation as padded 16x16 WMMA GEMMs; rows staged
//          into LDS by the Tensor Data Mover (double-buffered, s_wait_tensorcnt).
// Stage 2: tiny entropy epilogue (1 workgroup).

#define PADW  7
#define TS    15            // 2*PADW+1 shifts per axis
#define KCH   10            // channels (k)
#define BN    16            // batch
#define HH    224
#define WW    224
#define PITCH 268           // LDS halves/row; (PITCH/2)%64==6 -> conflict-free banks
#define GP    256           // packed global row pitch (halves)
#define YPAD  (HH + 2*PADW) // 238 padded rows in A16
#define WAVES 4
#define EPSF  1e-16f

typedef __attribute__((ext_vector_type(16))) _Float16     v16h;
typedef __attribute__((ext_vector_type(8)))  float        v8f;
typedef __attribute__((ext_vector_type(4)))  unsigned int u32x4;
typedef __attribute__((ext_vector_type(8)))  int          i32x8;
typedef __attribute__((ext_vector_type(4)))  int          i32x4;

union Frag16 { unsigned int u[8]; uint2 u2[4]; v16h h; };
union Pack8  { _Float16 h[8]; uint4 u4; };

// ---------------- Stage 0: pack to f16 with halo ----------------

__global__ __launch_bounds__(256) void pack_a_kernel(
    const float* __restrict__ xo, _Float16* __restrict__ A16)
{
  const int idx8 = blockIdx.x * 256 + threadIdx.x;  // one 8-half (16B) chunk
  const int xp8  = idx8 & (GP / 8 - 1);
  int t = idx8 >> 5;
  const int yp = t % YPAD; t /= YPAD;
  const int n  = t % KCH;  t /= KCH;
  const int c  = t;
  if (c >= BN) return;
  const int  y     = yp - PADW;
  const bool rowOK = (y >= 0) && (y < HH);
  const float* src = xo + (((size_t)c * KCH + n) * HH + (rowOK ? y : 0)) * WW;
  Pack8 pk;
#pragma unroll
  for (int j = 0; j < 8; ++j) {
    const int xg = xp8 * 8 + j - PADW;
    float v = 0.0f;
    if (rowOK && xg >= 0 && xg < WW) v = src[xg];
    pk.h[j] = (_Float16)v;
  }
  *(uint4*)(A16 + (size_t)idx8 * 8) = pk.u4;
}

__global__ __launch_bounds__(256) void pack_b_kernel(
    const float* __restrict__ xt, _Float16* __restrict__ B16)
{
  const int idx8 = blockIdx.x * 256 + threadIdx.x;
  const int xp8  = idx8 & (GP / 8 - 1);
  int t = idx8 >> 5;
  const int y = t % HH;  t /= HH;
  const int o = t % KCH; t /= KCH;
  const int c = t;
  if (c >= BN) return;
  const float* src = xt + (((size_t)c * KCH + o) * HH + y) * WW;
  Pack8 pk;
#pragma unroll
  for (int j = 0; j < 8; ++j) {
    const int xg = xp8 * 8 + j;
    float v = 0.0f;
    if (xg < WW) v = src[xg];
    pk.h[j] = (_Float16)v;
  }
  *(uint4*)(B16 + (size_t)idx8 * 8) = pk.u4;
}

// ---------------- TDM: 2D tile (10 rows x 256 halves) -> LDS, pitch 268 ----------------
// D# group1: data_size=1 (2B); pad_enable, pad_interval=6 (pad after 64x8B = 256 halves),
// pad_amount=5 (6 dwords = 12 halves) => LDS row pitch 256+12 = 268 halves.

__device__ __forceinline__ void tdm_load_rows(unsigned ldsOff,
                                              const _Float16* gptr,
                                              unsigned rowStrideElems)
{
  const unsigned long long ga = (unsigned long long)(uintptr_t)gptr;
  u32x4 g0;
  g0.x = 1u;                                            // count=1, user mode
  g0.y = ldsOff;                                        // LDS byte address
  g0.z = (unsigned)(ga & 0xFFFFFFFFu);                  // global addr lo
  g0.w = (unsigned)((ga >> 32) & 0x01FFFFFFu) | (2u << 30); // addr hi | type=2
  i32x8 g1;
  g1.s0 = (1 << 16) | (1 << 20) | (6 << 22) | (5 << 25); // data_size|pad_en|intvl|amt
  g1.s1 = (int)((unsigned)GP << 16);                    // tensor_dim0 lo16 (=256)
  g1.s2 = (int)((unsigned)KCH << 16);                   // tensor_dim0 hi | tensor_dim1 lo (=10)
  g1.s3 = (int)((unsigned)GP << 16);                    // tensor_dim1 hi | tile_dim0 (=256)
  g1.s4 = KCH;                                          // tile_dim1=10, tile_dim2=0
  g1.s5 = (int)rowStrideElems;                          // tensor_dim0_stride lo32
  g1.s6 = 0;                                            // stride0 hi | stride1 lo
  g1.s7 = 0;
  const i32x4 z4 = {0, 0, 0, 0};
  const i32x8 z8 = {0, 0, 0, 0, 0, 0, 0, 0};
  __builtin_amdgcn_tensor_load_to_lds(g0, g1, z4, z4, z8, 0);
}

// ---------------- Stage 1: correlation via WMMA ----------------

__global__ __launch_bounds__(WAVES * 32) void corr_wmma_kernel(
    const _Float16* __restrict__ A16, const _Float16* __restrict__ B16,
    float* __restrict__ p)
{
  extern __shared__ _Float16 smem[];     // 2 bufs * WAVES * 2 arrays * 16 * PITCH halves
  __shared__ float pLds[TS * 256];

  const int dy     = blockIdx.x;         // 0..14
  const int c      = blockIdx.y;         // 0..15
  const int strip  = blockIdx.z;         // 0..13
  const int tid    = threadIdx.x;
  const int w      = tid >> 5;
  const int lane   = tid & 31;
  const int laneHi = lane >> 4;
  const int mrow   = lane & 15;

  for (int i = tid; i < TS * 256; i += WAVES * 32) pLds[i] = 0.0f;
  __syncthreads();

  _Float16* wbase = smem + (size_t)w * (2 * 2 * 16 * PITCH);
  _Float16* bufA[2]; _Float16* bufB[2];
  bufA[0] = wbase;                      bufB[0] = bufA[0] + 16 * PITCH;
  bufA[1] = wbase + 2 * 16 * PITCH;     bufB[1] = bufA[1] + 16 * PITCH;

  // zero the padded rows n/o = 10..15 once (TDM only writes rows 0..9)
  for (int b = 0; b < 2; ++b)
    for (int a = 0; a < 2; ++a) {
      _Float16* arr = a ? bufB[b] : bufA[b];
      for (int n = KCH; n < 16; ++n) {
        unsigned* rw = (unsigned*)(arr + n * PITCH);
        for (int i = lane; i < PITCH / 2; i += 32) rw[i] = 0u;
      }
    }

  const int y0 = strip * (4 * WAVES) + w * 4;
  // issue row 0 into buffer 0
  tdm_load_rows((unsigned)(uintptr_t)bufA[0],
                A16 + ((size_t)c * KCH * YPAD + (y0 + dy)) * GP, YPAD * GP);
  tdm_load_rows((unsigned)(uintptr_t)bufB[0],
                B16 + ((size_t)c * KCH * HH + y0) * GP, HH * GP);

  const v8f vzero = {0.f, 0.f, 0.f, 0.f, 0.f, 0.f, 0.f, 0.f};
  v8f acc[TS];
#pragma unroll
  for (int d = 0; d < TS; ++d) acc[d] = vzero;

  for (int r = 0; r < 4; ++r) {
    const int cur = r & 1;
    if (r < 3) {
      // next buffer's previous reads must be retired before TDM overwrites it
      asm volatile("s_wait_dscnt 0x0" ::: "memory");
      const int y = y0 + r + 1;
      tdm_load_rows((unsigned)(uintptr_t)bufA[cur ^ 1],
                    A16 + ((size_t)c * KCH * YPAD + (y + dy)) * GP, YPAD * GP);
      tdm_load_rows((unsigned)(uintptr_t)bufB[cur ^ 1],
                    B16 + ((size_t)c * KCH * HH + y) * GP, HH * GP);
      __builtin_amdgcn_s_wait_tensorcnt(2);  // current row's 2 TDM ops done (in-order)
    } else {
      __builtin_amdgcn_s_wait_tensorcnt(0);
    }
    asm volatile("" ::: "memory");

    const _Float16* A = bufA[cur];
    const _Float16* B = bufB[cur];

    for (int xb = 0; xb < 7; ++xb) {
      const int base = xb * 32;
      // B fragment: lane = column o; runs K = 8*laneHi+{0..7} and 16+8*laneHi+{0..7}
      Frag16 bf;
      {
        const _Float16* bp = B + mrow * PITCH + base + 8 * laneHi;
        bf.u2[0] = *(const uint2*)(bp);
        bf.u2[1] = *(const uint2*)(bp + 4);
        bf.u2[2] = *(const uint2*)(bp + 16);
        bf.u2[3] = *(const uint2*)(bp + 20);
      }
      // A fragment (dx=0): runs K = 8*laneHi+{0..7} and 16+8*laneHi+{0..7}
      const _Float16* ap = A + mrow * PITCH + base + 8 * laneHi;
      Frag16 af;
      af.u2[0] = *(const uint2*)(ap);
      af.u2[1] = *(const uint2*)(ap + 4);
      af.u2[2] = *(const uint2*)(ap + 16);
      af.u2[3] = *(const uint2*)(ap + 20);

#pragma unroll
      for (int dx = 0; dx < TS; ++dx) {
        if (dx > 0) {
          const unsigned int n0 = *(const unsigned short*)(ap + 8  + (dx - 1));
          const unsigned int n1 = *(const unsigned short*)(ap + 24 + (dx - 1));
          af.u[0] = (af.u[0] >> 16) | (af.u[1] << 16);
          af.u[1] = (af.u[1] >> 16) | (af.u[2] << 16);
          af.u[2] = (af.u[2] >> 16) | (af.u[3] << 16);
          af.u[3] = (af.u[3] >> 16) | (n0 << 16);
          af.u[4] = (af.u[4] >> 16) | (af.u[5] << 16);
          af.u[5] = (af.u[5] >> 16) | (af.u[6] << 16);
          af.u[6] = (af.u[6] >> 16) | (af.u[7] << 16);
          af.u[7] = (af.u[7] >> 16) | (n1 << 16);
        }
        acc[dx] = __builtin_amdgcn_wmma_f32_16x16x32_f16(
            false, af.h, false, bf.h, (short)0, acc[dx], false, false);
      }
    }
  }

  // wave -> LDS (ds float atomics): C/D layout: lane VGPR i -> (M=i+8*laneHi, N=mrow)
#pragma unroll
  for (int dx = 0; dx < TS; ++dx) {
#pragma unroll
    for (int i = 0; i < 8; ++i) {
      const int m = i + 8 * laneHi;
      atomicAdd(&pLds[dx * 256 + m * 16 + mrow], acc[dx][i]);
    }
  }
  __syncthreads();

  float* pg = p + (size_t)dy * TS * 256;
  for (int i = tid; i < TS * 256; i += WAVES * 32) {
    unsafeAtomicAdd(&pg[i], pLds[i]);   // global_atomic_add_f32
  }
}

// ---------------- Stage 2: entropy epilogue ----------------

__global__ __launch_bounds__(256) void iid_loss_kernel(
    const float* __restrict__ p, float* __restrict__ out)
{
  __shared__ float red[256];
  const int  tid    = threadIdx.x;
  const bool active = tid < TS * TS;

  float q[KCH * KCH];
  float lmin = __builtin_inff();
  if (active) {
    const float* ps = p + (size_t)tid * 256;
    for (int i = 0; i < KCH; ++i)
      for (int j = 0; j < KCH; ++j) {
        const float v = ps[i * 16 + j];
        q[i * KCH + j] = v;
        lmin = fminf(lmin, v);
      }
  }
  red[tid] = lmin;
  __syncthreads();
  for (int s = 128; s > 0; s >>= 1) {
    if (tid < s) red[tid] = fminf(red[tid], red[tid + s]);
    __syncthreads();
  }
  const float gmin = red[0];
  __syncthreads();

  float lsum = 0.0f;
  if (active) {
    float tot = 0.0f;
    for (int i = 0; i < KCH * KCH; ++i) { q[i] = q[i] - gmin + EPSF; tot += q[i]; }
    const float inv = 1.0f / tot;
    for (int i = 0; i < KCH * KCH; ++i) q[i] *= inv;
    for (int i = 0; i < KCH; ++i)
      for (int j = 0; j < i; ++j) {
        const float s2 = 0.5f * (q[i * KCH + j] + q[j * KCH + i]);
        q[i * KCH + j] = s2; q[j * KCH + i] = s2;
      }
    float pi_[KCH], pj_[KCH];
    for (int j = 0; j < KCH; ++j) { pi_[j] = 0.f; pj_[j] = 0.f; }
    for (int i = 0; i < KCH; ++i)
      for (int j = 0; j < KCH; ++j) { pj_[i] += q[i * KCH + j]; pi_[j] += q[i * KCH + j]; }
    for (int i = 0; i < KCH; ++i)
      for (int j = 0; j < KCH; ++j) {
        const float v = q[i * KCH + j];
        lsum += -v * (logf(v + EPSF) - logf(pi_[j] + EPSF) - logf(pj_[i] + EPSF));
      }
  }
  red[tid] = lsum;
  __syncthreads();
  for (int s = 128; s > 0; s >>= 1) {
    if (tid < s) red[tid] += red[tid + s];
    __syncthreads();
  }
  if (tid == 0) out[0] = red[0] / (float)(TS * TS);
}

// ---------------- launch ----------------

extern "C" void kernel_launch(void* const* d_in, const int* in_sizes, int n_in,
                              void* d_out, int out_size, void* d_ws, size_t ws_size,
                              hipStream_t stream)
{
  (void)in_sizes; (void)n_in; (void)out_size; (void)ws_size;
  const float* xo = (const float*)d_in[0];
  const float* xt = (const float*)d_in[1];

  char* ws = (char*)d_ws;
  float*    p   = (float*)ws;                              // 230,400 B
  _Float16* A16 = (_Float16*)(ws + (1u << 20));            // 16*10*238*256*2 = 19,496,960 B
  _Float16* B16 = (_Float16*)(ws + (21u << 20));           // 16*10*224*256*2 = 18,350,080 B

  (void)hipMemsetAsync(p, 0, (size_t)TS * TS * 256 * sizeof(float), stream);

  const int aChunks = BN * KCH * YPAD * (GP / 8);          // 16B chunks
  const int bChunks = BN * KCH * HH   * (GP / 8);
  pack_a_kernel<<<(aChunks + 255) / 256, 256, 0, stream>>>(xo, A16);
  pack_b_kernel<<<(bChunks + 255) / 256, 256, 0, stream>>>(xt, B16);

  dim3 grid(TS, BN, HH / (4 * WAVES));                     // (15, 16, 14)
  const size_t shmem = (size_t)2 * WAVES * 2 * 16 * PITCH * sizeof(_Float16); // 137,216 B
  corr_wmma_kernel<<<grid, WAVES * 32, shmem, stream>>>(A16, B16, p);
  iid_loss_kernel<<<1, 256, 0, stream>>>(p, (float*)d_out);
}
